// sinLayer_90623809946047
// MI455X (gfx1250) — matrix operand order repaired and checked by reading, not compiled, not run on hardware
//
#include <hip/hip_runtime.h>
#include <cstdint>
#include <cstddef>

typedef _Float16 half_t;
typedef __attribute__((ext_vector_type(16))) _Float16 v16h;
typedef __attribute__((ext_vector_type(8)))  _Float16 v8h;
typedef __attribute__((ext_vector_type(8)))  float    v8f;

#define S_DETn 725
#define S_ANGn 360
#define OUTSZ  512
#define NNZn   8388608
#define BATCHn 4
#define CH     64
#define BN_EPSf 1e-3f

union AB16 { v16h v; v8h h[2]; };

// ---------------------------------------------------------------- utilities
__global__ void zero_f32(float* __restrict__ p, long n) {
    long i = (long)blockIdx.x * 256 + threadIdx.x;
    if (i < n) p[i] = 0.f;
}

// Convert one layer of 5x5x64x64 HWIO f32 weights into f16 [tap][co][ci]
__global__ void wcvt(const float* __restrict__ w, half_t* __restrict__ o) {
    int i = blockIdx.x * 256 + threadIdx.x;            // 25*64*64 = 102400
    if (i >= 25 * CH * CH) return;
    int ci = i & 63; int co = (i >> 6) & 63; int t = i >> 12;
    o[(t * CH + co) * CH + ci] = (half_t)w[((long)t * CH + ci) * CH + co];
}

// ------------------------------------------------- conv 5x5 1->64 + bias + ReLU
// writes f16 activations and initializes the f32 residual accumulator P
__global__ void conv_first(const float* __restrict__ in, const float* __restrict__ w,
                           const float* __restrict__ bias, half_t* __restrict__ act,
                           float* __restrict__ P, int H, int W) {
    __shared__ float wS[25 * CH];
    __shared__ float bS[CH];
    for (int e = threadIdx.x; e < 25 * CH; e += 256) wS[e] = w[e];
    if (threadIdx.x < CH) bS[threadIdx.x] = bias[threadIdx.x];
    __syncthreads();
    long t = (long)blockIdx.x * 256 + threadIdx.x;
    long total = (long)BATCHn * H * W * 4;
    if (t >= total) return;
    int q = (int)(t & 3); long pix = t >> 2;
    int c = (int)(pix % W); long rb = pix / W; int r = (int)(rb % H); int b = (int)(rb / H);
    float acc[16];
    #pragma unroll
    for (int j = 0; j < 16; j++) acc[j] = 0.f;
    for (int tr = 0; tr < 5; tr++) {
        int rr = r + tr - 2; if (rr < 0 || rr >= H) continue;
        for (int tc = 0; tc < 5; tc++) {
            int cc = c + tc - 2; if (cc < 0 || cc >= W) continue;
            float x = in[((long)b * H + rr) * W + cc];
            const float* wp = &wS[(tr * 5 + tc) * CH + q * 16];
            #pragma unroll
            for (int j = 0; j < 16; j++) acc[j] += x * wp[j];
        }
    }
    long base = pix * CH + q * 16;
    #pragma unroll
    for (int j = 0; j < 16; j++) {
        float y = acc[j] + bS[q * 16 + j];
        y = y > 0.f ? y : 0.f;
        act[base + j] = (half_t)y;
        P[base + j]   = y;
    }
}

// ---------------------------------------- WMMA implicit-GEMM conv 5x5 64->64
// + fused BatchNorm + ReLU + residual accumulate (P += y), f16 out for chain.
// Block = 256 thr = 8 waves; tile = 16x4 pixels (64) x 64 out channels.
// Each wave: 2 M-tiles (shared B fragment) x 1 N-tile; B loaded straight
// from L2 (weights are 200 KB/layer, grid-wide reuse) - no weight LDS, and
// only ONE barrier per block (after input staging).
__global__ void __launch_bounds__(256) conv_mid_wmma(
        const half_t* __restrict__ actIn, const half_t* __restrict__ wgt,
        half_t* __restrict__ actOut, float* __restrict__ P,
        const float* __restrict__ gamma, const float* __restrict__ beta,
        const float* __restrict__ mean,  const float* __restrict__ var,
        int H, int W) {
    __shared__ __align__(16) half_t inTile[20 * 8 * CH];   // 20 KB, [r][c][ci]
    const int tid  = threadIdx.x;
    const int b    = blockIdx.z;
    const int row0 = blockIdx.y * 16;
    const int col0 = blockIdx.x * 4;

    // Stage input tile with halo (zero pad); 1280 x v8h slots, 5 per thread
    for (int slot = tid; slot < 20 * 8 * CH / 8; slot += 256) {
        int r20 = slot >> 6; int rem = slot & 63; int c8 = rem >> 3; int ci8 = (rem & 7) * 8;
        int row = row0 + r20 - 2; int col = col0 + c8 - 2;
        v8h val = {};
        if (row >= 0 && row < H && col >= 0 && col < W)
            val = *(const v8h*)(actIn + (((size_t)b * H + row) * W + col) * CH + ci8);
        *(v8h*)(inTile + slot * 8) = val;
    }
    __syncthreads();

    const int wave = tid >> 5, lane = tid & 31;
    const int nT = wave & 3;                       // out-channel tile
    const int mT0 = (wave >> 2) * 2;               // first of 2 pixel tiles
    const int halfs = lane >> 4, l15 = lane & 15;
    // A-fragment source pixel for M-tile 0 (tile 1 is +16 pixels = +4 rows)
    const int p0 = mT0 * 16 + l15;
    const int pr = p0 >> 2, pc = p0 & 3;
    // B fragment global source: wgt[tap][co][ci], 32 contiguous halves per lane
    const half_t* bSrc = wgt + ((size_t)(nT * 16 + l15)) * CH + halfs * 16;

    v8f c0 = {}, c1 = {};
    for (int tap = 0; tap < 25; tap++) {
        const int tr = tap / 5, tc = tap % 5;
        const half_t* aBase = inTile + ((pr + tr) * 8 + (pc + tc)) * CH;
        const half_t* bTap  = bSrc + (size_t)tap * CH * CH;
        #pragma unroll
        for (int kc = 0; kc < CH; kc += 32) {
            AB16 a0, a1, bm;
            // A 16x32 f16 layout: VGPR0-3: K = kc + half*8 + 0..7
            //                     VGPR4-7: K = kc + 16 + half*8 + 0..7
            a0.h[0] = *(const v8h*)(aBase + kc + halfs * 8);
            a0.h[1] = *(const v8h*)(aBase + kc + 16 + halfs * 8);
            a1.h[0] = *(const v8h*)(aBase + 2048 + kc + halfs * 8);       // +16 px = +4 rows
            a1.h[1] = *(const v8h*)(aBase + 2048 + kc + 16 + halfs * 8);
            // B 32x16 f16 layout: VGPR0-3: K = kc + half*8 + 0..7, VGPR4-7: K = kc + 16 + half*8 + 0..7, N=l15
            bm.h[0] = *(const v8h*)(bTap - halfs * 8 + kc);
            bm.h[1] = *(const v8h*)(bTap - halfs * 8 + kc + 16);
            c0 = __builtin_amdgcn_wmma_f32_16x16x32_f16(
                     false, a0.v, false, bm.v, (short)0, c0, false, false);
            c1 = __builtin_amdgcn_wmma_f32_16x16x32_f16(
                     false, a1.v, false, bm.v, (short)0, c1, false, false);
        }
    }

    // Epilogue: BN scale/bias + ReLU + residual accumulate + f16 requant
    const int co = nT * 16 + l15;
    float s  = gamma[co] * rsqrtf(var[co] + BN_EPSf);
    float tt = beta[co] - mean[co] * s;
    #pragma unroll
    for (int i = 0; i < 8; i++) {
        int m = (halfs ? 8 : 0) + i;               // C layout: lane<16 M=i else M=8+i
        #pragma unroll
        for (int tile = 0; tile < 2; tile++) {
            int pp = (mT0 + tile) * 16 + m;
            int r = row0 + (pp >> 2), cl = col0 + (pp & 3);
            if (r < H && cl < W) {
                size_t idx = (((size_t)b * H + r) * W + cl) * CH + co;
                float y = (tile ? c1[i] : c0[i]) * s + tt;
                y = y > 0.f ? y : 0.f;
                P[idx] += y;
                actOut[idx] = (half_t)y;
            }
        }
    }
}

// ------------------------------------- conv 5x5 64->1 of (P * invdiv) + bias + skip
__global__ void conv_last(const float* __restrict__ P, const float* __restrict__ w,
                          const float* __restrict__ bias, const float* __restrict__ skip,
                          float* __restrict__ out, float invdiv, int H, int W) {
    __shared__ float wS[25 * CH];
    for (int e = threadIdx.x; e < 25 * CH; e += 256) wS[e] = w[e];
    __syncthreads();
    long t = (long)blockIdx.x * 256 + threadIdx.x;
    if (t >= (long)BATCHn * H * W) return;
    int c = (int)(t % W); long rb = t / W; int r = (int)(rb % H); int b = (int)(rb / H);
    float acc = 0.f;
    for (int tr = 0; tr < 5; tr++) {
        int rr = r + tr - 2; if (rr < 0 || rr >= H) continue;
        for (int tc = 0; tc < 5; tc++) {
            int cc = c + tc - 2; if (cc < 0 || cc >= W) continue;
            const float* pp = P + (((size_t)b * H + rr) * W + cc) * CH;
            const float* wp = &wS[(tr * 5 + tc) * CH];
            float a2 = 0.f;
            #pragma unroll 8
            for (int ci = 0; ci < CH; ci++) a2 += pp[ci] * wp[ci];
            acc += a2;
        }
    }
    out[t] = acc * invdiv + bias[0] + skip[t];
}

// ------------------------------------------------ 725-tap ramp filter (SAME)
// input de_sin is [B, det, ang]; output filt is [B, ang, det]
__global__ void ramp_filter(const float* __restrict__ sino, const float* __restrict__ wb,
                            float* __restrict__ filt) {
    __shared__ float xr[S_DETn];
    __shared__ float wr[S_DETn];
    int b = blockIdx.y, a = blockIdx.x;
    for (int d = threadIdx.x; d < S_DETn; d += 256) {
        xr[d] = sino[((long)b * S_DETn + d) * S_ANGn + a];
        wr[d] = wb[d];
    }
    __syncthreads();
    for (int d = threadIdx.x; d < S_DETn; d += 256) {
        float acc = 0.f;
        int k0 = 362 - d;               if (k0 < 0) k0 = 0;
        int k1 = S_DETn - 1 + 362 - d;  if (k1 > S_DETn - 1) k1 = S_DETn - 1;
        for (int k = k0; k <= k1; k++) acc += xr[d - 362 + k] * wr[k];
        filt[((long)b * S_ANGn + a) * S_DETn + d] = acc;
    }
}

// ------------------------------------------------ sparse backprojection scatter
__global__ void backproj(const float* __restrict__ filt, const float* __restrict__ vals,
                         const int* __restrict__ rows, const int* __restrict__ cols,
                         float* __restrict__ fbp) {
    long n = (long)blockIdx.x * 256 + threadIdx.x;
    if (n >= NNZn) return;
    int r = rows[n], c = cols[n];
    float v = 4.0f * vals[n];
    #pragma unroll
    for (int b = 0; b < BATCHn; b++)
        atomicAdd(&fbp[(long)b * OUTSZ * OUTSZ + r], v * filt[(long)b * S_DETn * S_ANGn + c]);
}

// ============================================================================
extern "C" void kernel_launch(void* const* d_in, const int* in_sizes, int n_in,
                              void* d_out, int out_size, void* d_ws, size_t ws_size,
                              hipStream_t stream) {
    const float* inputs    = (const float*)d_in[0];
    const float* w_sin1    = (const float*)d_in[1];
    const float* b_sin1    = (const float*)d_in[2];
    const float* w_sin_mid = (const float*)d_in[3];
    const float* sin_gamma = (const float*)d_in[4];
    const float* sin_beta  = (const float*)d_in[5];
    const float* sin_mean  = (const float*)d_in[6];
    const float* sin_var   = (const float*)d_in[7];
    const float* w_sin6    = (const float*)d_in[8];
    const float* b_sin6    = (const float*)d_in[9];
    const float* w_ct1     = (const float*)d_in[10];
    const float* b_ct1     = (const float*)d_in[11];
    const float* w_ct_mid  = (const float*)d_in[12];
    const float* ct_gamma  = (const float*)d_in[13];
    const float* ct_beta   = (const float*)d_in[14];
    const float* ct_mean   = (const float*)d_in[15];
    const float* ct_var    = (const float*)d_in[16];
    const float* w_ct6     = (const float*)d_in[17];
    const float* b_ct6     = (const float*)d_in[18];
    const float* w_b       = (const float*)d_in[19];
    const float* at_vals   = (const float*)d_in[20];
    const int*   at_rows   = (const int*)d_in[21];
    const int*   at_cols   = (const int*)d_in[22];

    float* out     = (float*)d_out;
    float* de_sin  = out;                                        // [4,725,360,1]
    float* outputs = out + (long)BATCHn * S_DETn * S_ANGn;       // [4,512,512,1]
    float* fbp     = outputs + (long)BATCHn * OUTSZ * OUTSZ;     // [4,512,512,1]

    // workspace layout (sized by the larger image-domain plane, 512*512)
    char* ws = (char*)d_ws;
    const size_t NPIX = (size_t)OUTSZ * OUTSZ;                   // 262144 >= 725*360
    half_t* actA = (half_t*)ws;                                              // 134 MB
    half_t* actB = (half_t*)(ws + (size_t)BATCHn * NPIX * CH * 2);           // 134 MB
    float*  P    = (float*) (ws + (size_t)BATCHn * NPIX * CH * 4);           // 268 MB
    half_t* wf16 = (half_t*)(ws + (size_t)BATCHn * NPIX * CH * 8);           // 200 KB
    float*  filt = (float*) (ws + (size_t)BATCHn * NPIX * CH * 8 + (1 << 20));

    // ---------------- sinogram-domain CNN ----------------
    int H = S_DETn, W = S_ANGn;
    long npx = (long)BATCHn * H * W;
    conv_first<<<(unsigned)((npx * 4 + 255) / 256), 256, 0, stream>>>(
        inputs, w_sin1, b_sin1, actA, P, H, W);
    dim3 gmid((W + 3) / 4, (H + 15) / 16, BATCHn);
    for (int i = 0; i < 4; i++) {
        wcvt<<<(25 * CH * CH + 255) / 256, 256, 0, stream>>>(
            w_sin_mid + (size_t)i * 25 * CH * CH, wf16);
        conv_mid_wmma<<<gmid, 256, 0, stream>>>(
            (i & 1) ? actB : actA, wf16, (i & 1) ? actA : actB, P,
            sin_gamma + i * CH, sin_beta + i * CH, sin_mean + i * CH, sin_var + i * CH, H, W);
    }
    conv_last<<<(unsigned)((npx + 255) / 256), 256, 0, stream>>>(
        P, w_sin6, b_sin6, inputs, de_sin, 0.25f, H, W);

    // ---------------- filtered backprojection ----------------
    ramp_filter<<<dim3(S_ANGn, BATCHn), 256, 0, stream>>>(de_sin, w_b, filt);
    zero_f32<<<(unsigned)(((long)BATCHn * OUTSZ * OUTSZ + 255) / 256), 256, 0, stream>>>(
        fbp, (long)BATCHn * OUTSZ * OUTSZ);
    backproj<<<(NNZn + 255) / 256, 256, 0, stream>>>(filt, at_vals, at_rows, at_cols, fbp);

    // ---------------- image-domain CNN ----------------
    H = OUTSZ; W = OUTSZ;
    npx = (long)BATCHn * H * W;
    conv_first<<<(unsigned)((npx * 4 + 255) / 256), 256, 0, stream>>>(
        fbp, w_ct1, b_ct1, actA, P, H, W);
    dim3 gmid2((W + 3) / 4, (H + 15) / 16, BATCHn);
    for (int i = 0; i < 5; i++) {
        wcvt<<<(25 * CH * CH + 255) / 256, 256, 0, stream>>>(
            w_ct_mid + (size_t)i * 25 * CH * CH, wf16);
        conv_mid_wmma<<<gmid2, 256, 0, stream>>>(
            (i & 1) ? actB : actA, wf16, (i & 1) ? actA : actB, P,
            ct_gamma + i * CH, ct_beta + i * CH, ct_mean + i * CH, ct_var + i * CH, H, W);
    }
    conv_last<<<(unsigned)((npx + 255) / 256), 256, 0, stream>>>(
        P, w_ct6, b_ct6, fbp, outputs, 0.2f, H, W);
}
